// CrossAttnBlock_46084999086749
// MI455X (gfx1250) — matrix operand order repaired and checked by reading, not compiled, not run on hardware
//
#include <hip/hip_runtime.h>

// ---------------------------------------------------------------------------
// CDNA5 (gfx1250) cross-attention block.  wave32, WMMA bf16 (f32 accumulate),
// async global->LDS staging for A operands, fragment-major packed bf16 weights.
// ---------------------------------------------------------------------------

typedef __attribute__((ext_vector_type(16))) __bf16 v16bf;
typedef __attribute__((ext_vector_type(8)))  __bf16 v8bf;
typedef __attribute__((ext_vector_type(8)))  float  v8f;

#define WMMA_BF16(a, b, c) \
  __builtin_amdgcn_wmma_f32_16x16x32_bf16(false, (a), false, (b), (short)0, (c), false, false)

static constexpr int Bsz = 8;
static constexpr int Nq  = 1024;
static constexpr int Mk  = 1024;
static constexpr int H   = 8;

__device__ __forceinline__ float siluf(float x) { return x / (1.f + __expf(-x)); }

__device__ __forceinline__ void wait_async0() {
#if __has_builtin(__builtin_amdgcn_s_wait_asynccnt)
  __builtin_amdgcn_s_wait_asynccnt(0);
#else
  asm volatile("s_wait_asynccnt 0" ::: "memory");
#endif
}

// ---- WMMA fragment loaders (contiguous k order, K = g*16 + e) ------------------
// A (16x32 bf16): lane l: M=l&15, g=l>>4; element e <- K=g*16+e
// -> two aligned 16B reads per lane.
__device__ __forceinline__ v16bf load_a_lds(const __bf16* A, int lda, int lane) {
  int m = lane & 15, g = (lane >> 4) & 1;
  const __bf16* p = A + (size_t)m * lda + g * 16;
  v8bf lo = *(const v8bf*)(p);
  v8bf hi = *(const v8bf*)(p + 8);
  v16bf a;
#pragma unroll
  for (int i = 0; i < 8; ++i) { a[i] = lo[i]; a[i + 8] = hi[i]; }
  return a;
}

// generic A loader (f32 global source), element-wise with cvt
template <typename T>
__device__ __forceinline__ v16bf load_a_rm(const T* A, int lda, int lane) {
  int m = lane & 15, g = (lane >> 4) & 1;
  const T* row = A + (size_t)m * lda;
  v16bf a;
#pragma unroll
  for (int r = 0; r < 8; ++r) {
    int k0 = g * 16 + 2 * r;
    a[2 * r]     = (__bf16)(float)row[k0];
    a[2 * r + 1] = (__bf16)(float)row[k0 + 1];
  }
  return a;
}

// B (32x16, row-major B[k][n]): lane l: N=l&15, g=l>>4; element e <- K=g*16+e
template <typename T>
__device__ __forceinline__ v16bf load_b_rm(const T* Bm, int ldb, int lane) {
  int n = lane & 15, g = (lane >> 4) & 1;
  v16bf b;
#pragma unroll
  for (int r = 0; r < 8; ++r) {
    int k0 = g * 16 + 2 * r;
    b[2 * r]     = (__bf16)(float)Bm[(size_t)k0 * ldb + n];
    b[2 * r + 1] = (__bf16)(float)Bm[(size_t)(k0 + 1) * ldb + n];
  }
  return b;
}

// packed-B loader: fragment-major layout, lane's 16 values contiguous (32B)
__device__ __forceinline__ v16bf load_b_pk(const __bf16* pk, int lane) {
  const v8bf* p = (const v8bf*)(pk + (size_t)lane * 16);
  v8bf lo = p[0], hi = p[1];
  v16bf b;
#pragma unroll
  for (int i = 0; i < 8; ++i) { b[i] = lo[i]; b[i + 8] = hi[i]; }
  return b;
}

// B = K^T : element (k,n) = Ksrc[n*lds + k]  (contiguous-k float2 loads)
__device__ __forceinline__ v16bf load_bT(const float* Ksrc, int lds, int lane) {
  int n = lane & 15, g = (lane >> 4) & 1;
  const float* row = Ksrc + (size_t)n * lds;
  v16bf b;
#pragma unroll
  for (int r = 0; r < 8; ++r) {
    int k0 = g * 16 + 2 * r;
    b[2 * r]     = (__bf16)row[k0];
    b[2 * r + 1] = (__bf16)row[k0 + 1];
  }
  return b;
}

// A from concat([src | pe]); 32-wide chunks never split the boundary
__device__ __forceinline__ v16bf load_a_cat(const float* src, const float* pe, int rowbase,
                                            int Dfeat, int Denc, int kbase, int lane) {
  int m = lane & 15, g = (lane >> 4) & 1;
  int grow = rowbase + m;
  const float* srow = src + (size_t)grow * Dfeat;
  const float* prow = pe + (size_t)(grow & (Nq - 1)) * Denc;
  v16bf a;
#pragma unroll
  for (int r = 0; r < 8; ++r) {
    int k0 = kbase + g * 16 + 2 * r;
    float x0 = (k0 < Dfeat) ? srow[k0] : prow[k0 - Dfeat];
    float x1 = (k0 + 1 < Dfeat) ? srow[k0 + 1] : prow[k0 + 1 - Dfeat];
    a[2 * r]     = (__bf16)x0;
    a[2 * r + 1] = (__bf16)x1;
  }
  return a;
}

// ---- weight packer: W [K,N] f32 -> [K/32][N/16][32 lanes][16 elems] bf16 ---
// lane l (tile-local): n=l&15, g=l>>4 ; element e holds W[kt*32+g*16+e][nt*16+n]
__global__ __launch_bounds__(256) void pack_b_kernel(const float* __restrict__ W, int N,
                                                     __bf16* __restrict__ dst) {
  const int lane = threadIdx.x & 31, wave = threadIdx.x >> 5;
  const int tile = blockIdx.x * 8 + wave;
  const int ntc = N >> 4;
  const int kt = tile / ntc, nt = tile % ntc;
  const int n = lane & 15, g = (lane >> 4) & 1;
  const float* src = W + (size_t)(kt * 32 + g * 16) * N + nt * 16 + n;
  __bf16* d = dst + (size_t)tile * 512 + lane * 16;
#pragma unroll
  for (int e = 0; e < 16; ++e) d[e] = (__bf16)src[(size_t)e * N];
}

// ---- async global -> LDS staging of a 16 x 256 bf16 tile (padded rows) ----
static constexpr int CK  = 256;
static constexpr int CKP = CK + 8;   // 16B-multiple stride, bank-conflict free

__device__ __forceinline__ void stage_async_tile(__bf16* dst /*[16][CKP]*/,
                                                 const __bf16* src, int lda, int tid) {
  const int r  = tid >> 4;
  const int cb = (tid & 15) * 16;
  const char* gp = (const char*)(src + (size_t)r * lda);
  uint32_t lp = (uint32_t)(uintptr_t)(dst + (size_t)r * CKP);
#pragma unroll
  for (int i = 0; i < (CK * 2) / 256; ++i) {
    asm volatile("global_load_async_to_lds_b128 %0, %1, off"
                 :: "v"(lp + cb + i * 256), "v"(gp + cb + i * 256)
                 : "memory");
  }
}

// ---------------------------------------------------------------------------
// Fused projection: h = cat(src,pe)@W1 + b1 ; s = silu(LN(h)) ; out = s@W2+b2
// W1/W2 supplied in packed fragment-major bf16 form.
// ---------------------------------------------------------------------------
__global__ __launch_bounds__(256) void fused_proj_kernel(
    const float* __restrict__ src, const float* __restrict__ pe,
    const __bf16* __restrict__ pk_w1, const float* __restrict__ pe_b,
    const float* __restrict__ ln_g, const float* __restrict__ ln_b,
    const __bf16* __restrict__ pk_w2, const float* __restrict__ b2,
    int Dfeat, int Denc, int N2, float* __restrict__ outMain, __bf16* __restrict__ outV) {
  constexpr int HQP = 260;
  constexpr int HSP = 264;
  __shared__ float hq[16][HQP];
  __shared__ __align__(16) __bf16 hs[16][HSP];

  const int tid = threadIdx.x, lane = tid & 31, wave = tid >> 5;
  const int rowbase = blockIdx.x * 16;
  const int K1 = Dfeat + Denc;
  const int n = lane & 15, g = (lane >> 4) & 1;

  // --- stage 1: hq = cat @ W1 + b1  (N=256 -> 16 col tiles, packed B) ------
#pragma unroll
  for (int tt = 0; tt < 2; ++tt) {
    int ct = wave + tt * 8;
    v8f acc = {};
    for (int kc = 0; kc < K1; kc += 32) {
      v16bf a = load_a_cat(src, pe, rowbase, Dfeat, Denc, kc, lane);
      v16bf b = load_b_pk(pk_w1 + ((size_t)(kc >> 5) * 16 + ct) * 512, lane);
      acc = WMMA_BF16(a, b, acc);
    }
    int col = ct * 16 + n;
    float bb = pe_b[col];
#pragma unroll
    for (int r = 0; r < 8; ++r) hq[r + 8 * g][col] = acc[r] + bb;
  }
  __syncthreads();

  // --- stage 2: per-row LN + SiLU -> bf16 LDS ------------------------------
#pragma unroll
  for (int rr = 0; rr < 2; ++rr) {
    int row = wave + rr * 8;
    float vbuf[8], s = 0.f, s2 = 0.f;
#pragma unroll
    for (int i = 0; i < 8; ++i) {
      float v = hq[row][lane * 8 + i];
      vbuf[i] = v; s += v; s2 += v * v;
    }
#pragma unroll
    for (int msk = 16; msk >= 1; msk >>= 1) {
      s  += __shfl_xor(s,  msk, 32);
      s2 += __shfl_xor(s2, msk, 32);
    }
    float mean = s * (1.f / 256.f);
    float var  = s2 * (1.f / 256.f) - mean * mean;
    float rstd = rsqrtf(var + 1e-5f);
#pragma unroll
    for (int i = 0; i < 8; ++i) {
      int c = lane * 8 + i;
      float t = (vbuf[i] - mean) * rstd * ln_g[c] + ln_b[c];
      hs[row][c] = (__bf16)siluf(t);
    }
  }
  __syncthreads();

  // --- stage 3: out = hs @ W2 + b2 (A frags cached, packed B) --------------
  v16bf afr[8];
#pragma unroll
  for (int kc = 0; kc < 8; ++kc) afr[kc] = load_a_lds(&hs[0][kc * 32], HSP, lane);

  const int ntiles = N2 >> 4;
  for (int ct = wave; ct < ntiles; ct += 8) {
    v8f acc = {};
#pragma unroll
    for (int kc = 0; kc < 8; ++kc) {
      v16bf b = load_b_pk(pk_w2 + ((size_t)kc * ntiles + ct) * 512, lane);
      acc = WMMA_BF16(afr[kc], b, acc);
    }
    int col = ct * 16 + n;
    float bb = b2[col];
#pragma unroll
    for (int r = 0; r < 8; ++r) {
      int grow = rowbase + r + 8 * g;
      float val = acc[r] + bb;
      if (outV && col >= 512)
        outV[(size_t)grow * 2048 + (col - 512)] = (__bf16)val;
      else
        outMain[(size_t)grow * 512 + col] = val;
    }
  }
}

// ---- ||k||^2 per (row, head) ---------------------------------------------
__global__ __launch_bounds__(256) void norms64_kernel(const float* __restrict__ src,
                                                      float* __restrict__ dst) {
  int row = blockIdx.x, wave = threadIdx.x >> 5, lane = threadIdx.x & 31;
  const float* p = src + (size_t)row * 512 + wave * 64;
  float a = p[lane], b = p[lane + 32];
  float s = a * a + b * b;
#pragma unroll
  for (int msk = 16; msk >= 1; msk >>= 1) s += __shfl_xor(s, msk, 32);
  if (lane == 0) dst[row * 8 + wave] = s;
}

// ---------------------------------------------------------------------------
// Flash-style distance attention.
// ---------------------------------------------------------------------------
__global__ __launch_bounds__(256) void attn_kernel(
    const float* __restrict__ q_ws, const float* __restrict__ k_ws,
    const __bf16* __restrict__ v_ws, const float* __restrict__ ksq,
    const float* __restrict__ scale_p, __bf16* __restrict__ attn_out) {
  constexpr int PFP = 36, PBP = 40;
  __shared__ float Pf[16][PFP];
  __shared__ __align__(16) __bf16 Pb[16][PBP];
  __shared__ float qqs[16];
  __shared__ float den[16];

  const int tid = threadIdx.x, lane = tid & 31, wave = tid >> 5;
  const int nt = blockIdx.x, h = blockIdx.y, b = blockIdx.z;
  const int n = lane & 15, g = (lane >> 4) & 1;

  const float sc = scale_p[0];
  const float inv_s2 = 1.f / (sc * sc);

  const float* qbase = q_ws + (size_t)(b * Nq + nt * 16) * 512 + h * 64;
  if (tid < 16) {
    den[tid] = 0.f;
    const float* qr = qbase + (size_t)tid * 512;
    float s = 0.f;
#pragma unroll
    for (int d = 0; d < 64; ++d) s += qr[d] * qr[d];
    qqs[tid] = s;
  }
  v16bf aq0 = load_a_rm(qbase, 512, lane);
  v16bf aq1 = load_a_rm(qbase + 32, 512, lane);
  v8f accO[2] = {};
  __syncthreads();

  for (int mc = 0; mc < Mk / 32; ++mc) {
    const int mbase = mc * 32;
    if (wave < 2) {
      const int mt = mbase + wave * 16;
      const float* kb = k_ws + (size_t)(b * Mk + mt) * 512 + h * 64;
      v16bf bk0 = load_bT(kb, 512, lane);
      v16bf bk1 = load_bT(kb + 32, 512, lane);
      v8f s = {};
      s = WMMA_BF16(aq0, bk0, s);
      s = WMMA_BF16(aq1, bk1, s);
      int mg = mt + n;
      float kk = ksq[(size_t)(b * Mk + mg) * 8 + h];
#pragma unroll
      for (int r = 0; r < 8; ++r) {
        int row = r + 8 * g;
        float sq = fmaxf(qqs[row] + kk - 2.f * s[r], 0.f);
        float z = __expf(-sq * inv_s2);
        float p = __expf(z);
        Pf[row][wave * 16 + n] = p;
        Pb[row][wave * 16 + n] = (__bf16)p;
      }
    }
    __syncthreads();

    { // denominator (LDS float atomics -> ds_add_f32)
      int row = tid >> 4, c = (tid & 15) * 2;
      atomicAdd(&den[row], Pf[row][c] + Pf[row][c + 1]);
    }

    v16bf ap = load_a_lds(&Pb[0][0], PBP, lane);
#pragma unroll
    for (int t = 0; t < 2; ++t) {
      int c0 = (wave * 2 + t) * 16;
      v16bf bv = load_b_rm(v_ws + (size_t)(b * Mk + mbase) * 2048 + h * 256 + c0, 2048, lane);
      accO[t] = WMMA_BF16(ap, bv, accO[t]);
    }
    __syncthreads();
  }

#pragma unroll
  for (int t = 0; t < 2; ++t) {
    int c0 = (wave * 2 + t) * 16;
#pragma unroll
    for (int r = 0; r < 8; ++r) {
      int row = r + 8 * g;
      float val = accO[t][r] / den[row];
      attn_out[(size_t)(b * Nq + nt * 16 + row) * 2048 + h * 256 + c0 + n] = (__bf16)val;
    }
  }
}

// ---- per-row LN + SiLU over 256 (FFN input) -------------------------------
__global__ __launch_bounds__(256) void ln_silu_kernel(const float* __restrict__ src,
                                                      const float* __restrict__ gam,
                                                      const float* __restrict__ bet,
                                                      __bf16* __restrict__ dst) {
  int row = blockIdx.x * 8 + (threadIdx.x >> 5);
  int lane = threadIdx.x & 31;
  const float* p = src + (size_t)row * 256;
  float vbuf[8], s = 0.f, s2 = 0.f;
#pragma unroll
  for (int i = 0; i < 8; ++i) {
    float v = p[lane * 8 + i];
    vbuf[i] = v; s += v; s2 += v * v;
  }
#pragma unroll
  for (int msk = 16; msk >= 1; msk >>= 1) {
    s  += __shfl_xor(s,  msk, 32);
    s2 += __shfl_xor(s2, msk, 32);
  }
  float mean = s * (1.f / 256.f);
  float var  = s2 * (1.f / 256.f) - mean * mean;
  float rstd = rsqrtf(var + 1e-5f);
#pragma unroll
  for (int i = 0; i < 8; ++i) {
    int c = lane * 8 + i;
    float t = (vbuf[i] - mean) * rstd * gam[c] + bet[c];
    dst[(size_t)row * 256 + c] = (__bf16)siluf(t);
  }
}

// ---------------------------------------------------------------------------
// Generic WMMA GEMM: async double-buffered A staging + packed bf16 B.
// ---------------------------------------------------------------------------
template <typename DT, bool SILU, bool RESID, int MAXT>
__global__ __launch_bounds__(256) void gemm_ws_kernel(
    const __bf16* __restrict__ A, int lda, const __bf16* __restrict__ pkB,
    const float* __restrict__ bias, const float* __restrict__ resid, int ldr,
    DT* __restrict__ D, int ldd, int K) {
  __shared__ __align__(16) __bf16 As[2][16][CKP];
  const int tid = threadIdx.x, lane = tid & 31, wave = tid >> 5;
  const int rowbase = blockIdx.x * 16;
  const int n = lane & 15, g = (lane >> 4) & 1;
  const int nchunks = K / CK;
  constexpr int NT = MAXT * 8;  // column tiles
  v8f acc[MAXT];
#pragma unroll
  for (int t = 0; t < MAXT; ++t) acc[t] = (v8f){};

  stage_async_tile(&As[0][0][0], A + (size_t)rowbase * lda, lda, tid);
  wait_async0();
  __syncthreads();

  for (int c = 0; c < nchunks; ++c) {
    const int cur = c & 1;
    if (c + 1 < nchunks)
      stage_async_tile(&As[cur ^ 1][0][0],
                       A + (size_t)rowbase * lda + (size_t)(c + 1) * CK, lda, tid);
#pragma unroll
    for (int kc = 0; kc < CK; kc += 32) {
      v16bf a = load_a_lds(&As[cur][0][kc], CKP, lane);
      const size_t kt = (size_t)(c * CK + kc) >> 5;
#pragma unroll
      for (int t = 0; t < MAXT; ++t) {
        int ct = wave + t * 8;
        v16bf b = load_b_pk(pkB + (kt * NT + ct) * 512, lane);
        acc[t] = WMMA_BF16(a, b, acc[t]);
      }
    }
    wait_async0();
    __syncthreads();
  }

#pragma unroll
  for (int t = 0; t < MAXT; ++t) {
    int ct = wave + t * 8;
    int col = ct * 16 + n;
    float bb = bias[col];
#pragma unroll
    for (int r = 0; r < 8; ++r) {
      int grow = rowbase + r + 8 * g;
      float val = acc[t][r] + bb;
      if (SILU)  val = siluf(val);
      if (RESID) val += resid[(size_t)grow * ldr + col];
      D[(size_t)grow * ldd + col] = (DT)val;
    }
  }
}

// ---------------------------------------------------------------------------
extern "C" void kernel_launch(void* const* d_in, const int* in_sizes, int n_in,
                              void* d_out, int out_size, void* d_ws, size_t ws_size,
                              hipStream_t stream) {
  const float* x        = (const float*)d_in[0];
  const float* context  = (const float*)d_in[1];
  const float* q_pe     = (const float*)d_in[2];
  const float* q_pe_w   = (const float*)d_in[3];
  const float* q_pe_b   = (const float*)d_in[4];
  const float* q_ln_g   = (const float*)d_in[5];
  const float* q_ln_b   = (const float*)d_in[6];
  const float* q_w      = (const float*)d_in[7];
  const float* q_b      = (const float*)d_in[8];
  const float* kv_pe    = (const float*)d_in[9];
  const float* kv_pe_w  = (const float*)d_in[10];
  const float* kv_pe_b  = (const float*)d_in[11];
  const float* kv_ln_g  = (const float*)d_in[12];
  const float* kv_ln_b  = (const float*)d_in[13];
  const float* kv_w     = (const float*)d_in[14];
  const float* kv_b     = (const float*)d_in[15];
  const float* scale    = (const float*)d_in[16];
  const float* merge_w  = (const float*)d_in[17];
  const float* merge_b  = (const float*)d_in[18];
  const float* ff_ln_g  = (const float*)d_in[19];
  const float* ff_ln_b  = (const float*)d_in[20];
  const float* ff1_w    = (const float*)d_in[21];
  const float* ff1_b    = (const float*)d_in[22];
  const float* ff2_w    = (const float*)d_in[23];
  const float* ff2_b    = (const float*)d_in[24];

  const size_t ROWS = (size_t)Bsz * Nq;  // 8192

  char* ws = (char*)d_ws;
  float*  q_ws     = (float*)ws;   ws += ROWS * 512 * sizeof(float);
  float*  k_ws     = (float*)ws;   ws += ROWS * 512 * sizeof(float);
  __bf16* v_ws     = (__bf16*)ws;  ws += ROWS * 2048 * sizeof(__bf16);
  float*  ksq_ws   = (float*)ws;   ws += ROWS * 8 * sizeof(float);
  __bf16* attn_out = (__bf16*)ws;  ws += ROWS * 2048 * sizeof(__bf16);
  float*  x2_ws    = (float*)ws;   ws += ROWS * 256 * sizeof(float);
  __bf16* lnact_ws = (__bf16*)ws;  ws += ROWS * 256 * sizeof(__bf16);
  __bf16* h1_ws    = (__bf16*)ws;  ws += ROWS * 1024 * sizeof(__bf16);
  // packed weights (fragment-major bf16): tiles = (K/32)*(N/16), 512 elems each
  __bf16* pk_qpe   = (__bf16*)ws;  ws += (size_t)144  * 512 * sizeof(__bf16); // 288x256
  __bf16* pk_qw    = (__bf16*)ws;  ws += (size_t)256  * 512 * sizeof(__bf16); // 256x512
  __bf16* pk_kvpe  = (__bf16*)ws;  ws += (size_t)384  * 512 * sizeof(__bf16); // 768x256
  __bf16* pk_kvw   = (__bf16*)ws;  ws += (size_t)1280 * 512 * sizeof(__bf16); // 256x2560
  __bf16* pk_mrg   = (__bf16*)ws;  ws += (size_t)1024 * 512 * sizeof(__bf16); // 2048x256
  __bf16* pk_ff1   = (__bf16*)ws;  ws += (size_t)512  * 512 * sizeof(__bf16); // 256x1024
  __bf16* pk_ff2   = (__bf16*)ws;  ws += (size_t)512  * 512 * sizeof(__bf16); // 1024x256

  // ---- pack all weight matrices (one-time per launch, ~4 MB) --------------
  pack_b_kernel<<<144 / 8, 256, 0, stream>>>(q_pe_w,  256,  pk_qpe);
  pack_b_kernel<<<256 / 8, 256, 0, stream>>>(q_w,     512,  pk_qw);
  pack_b_kernel<<<384 / 8, 256, 0, stream>>>(kv_pe_w, 256,  pk_kvpe);
  pack_b_kernel<<<1280 / 8, 256, 0, stream>>>(kv_w,   2560, pk_kvw);
  pack_b_kernel<<<1024 / 8, 256, 0, stream>>>(merge_w, 256, pk_mrg);
  pack_b_kernel<<<512 / 8, 256, 0, stream>>>(ff1_w,   1024, pk_ff1);
  pack_b_kernel<<<512 / 8, 256, 0, stream>>>(ff2_w,   256,  pk_ff2);

  const int rowBlocks = (int)(ROWS / 16);  // 512

  fused_proj_kernel<<<rowBlocks, 256, 0, stream>>>(
      x, q_pe, pk_qpe, q_pe_b, q_ln_g, q_ln_b, pk_qw, q_b, 256, 32, 512, q_ws, nullptr);

  fused_proj_kernel<<<rowBlocks, 256, 0, stream>>>(
      context, kv_pe, pk_kvpe, kv_pe_b, kv_ln_g, kv_ln_b, pk_kvw, kv_b,
      256, 512, 2560, k_ws, v_ws);

  norms64_kernel<<<(int)ROWS, 256, 0, stream>>>(k_ws, ksq_ws);

  attn_kernel<<<dim3(Nq / 16, H, Bsz), 256, 0, stream>>>(
      q_ws, k_ws, v_ws, ksq_ws, scale, attn_out);

  // x2 = x + attn_out @ merge_w + b          (Ncols=256 -> MAXT=2, K=2048)
  gemm_ws_kernel<float, false, true, 2><<<rowBlocks, 256, 0, stream>>>(
      attn_out, 2048, pk_mrg, merge_b, x, 256, x2_ws, 256, 2048);

  // silu(LN(x2))
  ln_silu_kernel<<<(int)(ROWS / 8), 256, 0, stream>>>(x2_ws, ff_ln_g, ff_ln_b, lnact_ws);

  // h1 = silu(lnact @ ff1_w + b)             (Ncols=1024 -> MAXT=8, K=256)
  gemm_ws_kernel<__bf16, true, false, 8><<<rowBlocks, 256, 0, stream>>>(
      lnact_ws, 256, pk_ff1, ff1_b, nullptr, 0, h1_ws, 1024, 256);

  // out = x2 + h1 @ ff2_w + b                (Ncols=256 -> MAXT=2, K=1024)
  gemm_ws_kernel<float, false, true, 2><<<rowBlocks, 256, 0, stream>>>(
      h1_ws, 1024, pk_ff2, ff2_b, x2_ws, 256, (float*)d_out, 256, 1024);
}
